// MoELayer_20830591386091
// MI455X (gfx1250) — matrix operand order repaired and checked by reading, not compiled, not run on hardware
//
#include <hip/hip_runtime.h>
#include <hip/hip_bf16.h>
#include <math.h>

// ---------------------------------------------------------------------------
// MoE (sigmoid router, top-4, SwiGLU experts) for MI455X / gfx1250.
//
// Roofline: sparse expert work = T*K*3*2*D*H ~= 51.5 GFLOP. Weights (192 MB
// fp32) fit the 192 MB global L2, so the weight stream is L2-resident.
// Strategy: one-time repack of weights to bf16 in WMMA-fragment-major layout
// (96 MB in workspace, L2-resident), then v_wmma_f32_16x16x32_bf16 with
// M=32 token tiles (B-fragment reused across two M-subtiles -> 4 WMMAs per
// fragment pair). B loads become contiguous 32 B/lane (global_load_b128 x2).
// ---------------------------------------------------------------------------

#define T_TOK   2048
#define DIM     1024
#define HID     1024
#define NEXP    16
#define TOPK    4
#define EPSF    1e-6f

// fragment-major swizzle: per expert matrix (1024x1024),
// element offset = nt*16384 + kt*512 + lane*16 + i   (bf16 elements)
// with k = kt*32 + (lane>>4)*16 + i, n = nt*16 + (lane&15).
#define SWZ_EXPERT_ELEMS (1024 * 1024)           // 2 MB bf16 per expert matrix
#define SWZ_MAT_ELEMS    ((size_t)NEXP * SWZ_EXPERT_ELEMS)
#define SWZ_OFF_BYTES    ((size_t)1 << 20)       // stats live in first 1 MB
#define SWZ_BYTES        ((size_t)3 * SWZ_MAT_ELEMS * 2)

typedef __attribute__((ext_vector_type(16))) __bf16 v16bf;
typedef __attribute__((ext_vector_type(8)))  float  v8f;

// ---------------------------------------------------------------------------
// 0) zero output + per-expert counters (harness poisons, never re-zeros)
// ---------------------------------------------------------------------------
__global__ void zero_kernel(float* __restrict__ out, int n_out,
                            int* __restrict__ counts, float* __restrict__ ssum) {
    int i = blockIdx.x * blockDim.x + threadIdx.x;
    if (i < n_out) out[i] = 0.0f;
    if (i < NEXP) { counts[i] = 0; ssum[i] = 0.0f; }
}

// ---------------------------------------------------------------------------
// 1) Router: sigmoid -> normalize -> top-4 -> normalized combine weights.
// ---------------------------------------------------------------------------
__global__ __launch_bounds__(256)
void router_kernel(const float* __restrict__ x,
                   const float* __restrict__ gate_w,
                   const float* __restrict__ ebias,
                   int*   __restrict__ counts,
                   float* __restrict__ ssum,
                   int*   __restrict__ etok,
                   float* __restrict__ ewt) {
    const int t = blockIdx.x;
    __shared__ float part[16][NEXP];
    __shared__ float sc[NEXP];

    const int e = threadIdx.x & 15;
    const int g = threadIdx.x >> 4;
    float acc = 0.0f;
    const float* xr = x + (size_t)t * DIM;
    for (int d = g; d < DIM; d += 16)
        acc += xr[d] * gate_w[d * NEXP + e];
    part[g][e] = acc;
    __syncthreads();

    if (threadIdx.x < NEXP) {
        float l = ebias[threadIdx.x];
        #pragma unroll
        for (int j = 0; j < 16; ++j) l += part[j][threadIdx.x];
        sc[threadIdx.x] = 1.0f / (1.0f + __expf(-l));
    }
    __syncthreads();

    if (threadIdx.x == 0) {
        float s[NEXP], tot = 0.0f;
        #pragma unroll
        for (int i = 0; i < NEXP; ++i) tot += sc[i];
        const float inv = 1.0f / (tot + EPSF);
        #pragma unroll
        for (int i = 0; i < NEXP; ++i) {
            s[i] = sc[i] * inv;
            atomicAdd(&ssum[i], s[i]);
        }
        int   idx[TOPK];
        float val[TOPK];
        bool  used[NEXP] = {};
        float wsum = 0.0f;
        #pragma unroll
        for (int k = 0; k < TOPK; ++k) {
            int   bi = 0;
            float bv = -1.0f;
            #pragma unroll
            for (int i = 0; i < NEXP; ++i)
                if (!used[i] && s[i] > bv) { bv = s[i]; bi = i; }
            used[bi] = true; idx[k] = bi; val[k] = bv; wsum += bv;
        }
        const float invw = 1.0f / (wsum + EPSF);
        #pragma unroll
        for (int k = 0; k < TOPK; ++k) {
            const int ex   = idx[k];
            const int slot = atomicAdd(&counts[ex], 1);
            etok[ex * T_TOK + slot] = t;
            ewt [ex * T_TOK + slot] = val[k] * invw;
        }
    }
}

// ---------------------------------------------------------------------------
// 1b) Weight repack: fp32 [E][1024][1024] -> bf16 fragment-major swizzle.
//     One thread per (matrix, expert, ntile, ktile, lane): 16 elements.
// ---------------------------------------------------------------------------
__global__ __launch_bounds__(256)
void swizzle_weights_kernel(const float* __restrict__ w1,
                            const float* __restrict__ w3,
                            const float* __restrict__ w2,
                            __bf16* __restrict__ s1,
                            __bf16* __restrict__ s3,
                            __bf16* __restrict__ s2) {
    const unsigned gid = blockIdx.x * 256u + threadIdx.x;
    const int lane = gid & 31;
    const int kt   = (gid >> 5) & 31;
    const int nt   = (gid >> 10) & 63;
    const int e    = (gid >> 16) & 15;
    const int m    = gid >> 20;           // 0..2
    if (m >= 3) return;

    const float* src = (m == 0) ? w1 : (m == 1) ? w3 : w2;
    __bf16*      dst = (m == 0) ? s1 : (m == 1) ? s3 : s2;

    const int c  = lane >> 4;
    const int ln = lane & 15;
    const float* sp = src + (size_t)e * SWZ_EXPERT_ELEMS
                          + (size_t)(kt * 32 + c * 16) * 1024 + nt * 16 + ln;
    v16bf v;
    #pragma unroll
    for (int i = 0; i < 16; ++i) v[i] = (__bf16)sp[(size_t)i * 1024];

    __bf16* dp = dst + (size_t)e * SWZ_EXPERT_ELEMS
                     + (size_t)nt * 16384 + kt * 512 + lane * 16;
    *(v16bf*)dp = v;
}

// ---------------------------------------------------------------------------
// 2) Expert FFN: grid (expert, token-tile/32), 256 thr = 8 wave32.
//    LDS: bf16 x-tile [32][1024] + bf16 act tile [32][1024] = 128 KB.
//    SWZ=true: B fragments are contiguous bf16 (global_load_b128 x2).
//    SWZ=false: fallback strided fp32 loads + cvt (used if ws too small).
// ---------------------------------------------------------------------------
template <bool SWZ>
__global__ __launch_bounds__(256)
void expert_ffn_kernel(const float*  __restrict__ x,
                       const float*  __restrict__ w1f,
                       const float*  __restrict__ w3f,
                       const float*  __restrict__ w2f,
                       const __bf16* __restrict__ w1s,
                       const __bf16* __restrict__ w3s,
                       const __bf16* __restrict__ w2s,
                       const int*    __restrict__ counts,
                       const int*    __restrict__ etok,
                       const float*  __restrict__ ewt,
                       float*        __restrict__ out) {
    extern __shared__ __bf16 lds[];
    __bf16* xs = lds;                 // [32][1024] bf16 x tile
    __bf16* as = lds + 32 * DIM;      // [32][1024] bf16 activation tile

    const int e    = blockIdx.x;
    const int tile = blockIdx.y;
    const int cnt  = counts[e];
    if (tile * 32 >= cnt) return;
    const int base = e * T_TOK + tile * 32;

    // --- cooperative load of the gathered 32-token tile, fp32 -> bf16 ---
    {
        const int row  = threadIdx.x >> 3;           // 0..31
        const int col0 = (threadIdx.x & 7) * 128;    // 128 elems per thread
        const int ridx = tile * 32 + row;
        const int tok  = (ridx < cnt) ? etok[base + row] : 0;
        const float* src = x + (size_t)tok * DIM + col0;
        __bf16*      dst = xs + row * DIM + col0;
        #pragma unroll
        for (int i = 0; i < 128; i += 4) {
            float4 v = *(const float4*)(src + i);
            dst[i + 0] = (__bf16)v.x;
            dst[i + 1] = (__bf16)v.y;
            dst[i + 2] = (__bf16)v.z;
            dst[i + 3] = (__bf16)v.w;
        }
    }
    __syncthreads();

    const int lane = threadIdx.x & 31;
    const int wave = threadIdx.x >> 5;   // 0..7
    const int c    = lane >> 4;
    const int ln   = lane & 15;

    // ---------------- Phase 1: h/g tiles over HID, SiLU-gate ----------------
    for (int nt = wave; nt < HID / 16; nt += 8) {
        v8f hc0 = {}, hc1 = {}, gc0 = {}, gc1 = {};
        const int ncol = nt * 16 + ln;
        // fragment stream base (fast path)
        const __bf16* fp1 = w1s + (size_t)e * SWZ_EXPERT_ELEMS
                                + (size_t)nt * 16384 + lane * 16;
        const __bf16* fp3 = w3s + (size_t)e * SWZ_EXPERT_ELEMS
                                + (size_t)nt * 16384 + lane * 16;
        for (int k0 = 0; k0 < DIM; k0 += 32) {
            // A fragments: row = ln (+16), K = c*16 .. c*16+15 (contiguous, matches B)
            v16bf A0, A1;
            {
                const __bf16* a0 = xs + ln * DIM + k0 + c * 16;
                const __bf16* a1 = a0 + 16 * DIM;
                #pragma unroll
                for (int i = 0; i < 8; ++i) {
                    A0[i] = a0[i]; A0[i + 8] = a0[8 + i];
                    A1[i] = a1[i]; A1[i + 8] = a1[8 + i];
                }
            }
            v16bf B1, B2;
            if constexpr (SWZ) {
                B1 = *(const v16bf*)fp1;
                B2 = *(const v16bf*)fp3;
                __builtin_prefetch(fp1 + 2048, 0, 1);   // 4 fragments ahead
                __builtin_prefetch(fp3 + 2048, 0, 1);
                fp1 += 512;
                fp3 += 512;
            } else {
                const float* b1p = w1f + (size_t)e * SWZ_EXPERT_ELEMS
                                       + (size_t)(k0 + c * 16) * HID + ncol;
                const float* b3p = w3f + (size_t)e * SWZ_EXPERT_ELEMS
                                       + (size_t)(k0 + c * 16) * HID + ncol;
                #pragma unroll
                for (int i = 0; i < 16; ++i) {
                    B1[i] = (__bf16)b1p[(size_t)i * HID];
                    B2[i] = (__bf16)b3p[(size_t)i * HID];
                }
            }
            hc0 = __builtin_amdgcn_wmma_f32_16x16x32_bf16(false, A0, false, B1,
                                                          (short)0, hc0, false, false);
            hc1 = __builtin_amdgcn_wmma_f32_16x16x32_bf16(false, A1, false, B1,
                                                          (short)0, hc1, false, false);
            gc0 = __builtin_amdgcn_wmma_f32_16x16x32_bf16(false, A0, false, B2,
                                                          (short)0, gc0, false, false);
            gc1 = __builtin_amdgcn_wmma_f32_16x16x32_bf16(false, A1, false, B2,
                                                          (short)0, gc1, false, false);
        }
        // SwiGLU: a = silu(h)*g ; C/D layout: row m = r + 8*c (+16), col = ncol
        #pragma unroll
        for (int r = 0; r < 8; ++r) {
            const int m = r + 8 * c;
            float hv = hc0[r];
            as[m * HID + ncol] = (__bf16)((hv / (1.0f + __expf(-hv))) * gc0[r]);
            hv = hc1[r];
            as[(m + 16) * HID + ncol] = (__bf16)((hv / (1.0f + __expf(-hv))) * gc1[r]);
        }
    }
    __syncthreads();

    // ---------------- Phase 2: y = A @ W2, weighted scatter-add -------------
    for (int dt = wave; dt < DIM / 16; dt += 8) {
        v8f acc0 = {}, acc1 = {};
        const int ncol = dt * 16 + ln;
        const __bf16* fp2 = w2s + (size_t)e * SWZ_EXPERT_ELEMS
                                + (size_t)dt * 16384 + lane * 16;
        for (int k0 = 0; k0 < HID; k0 += 32) {
            v16bf A0, A1;
            {
                const __bf16* a0 = as + ln * HID + k0 + c * 16;
                const __bf16* a1 = a0 + 16 * HID;
                #pragma unroll
                for (int i = 0; i < 8; ++i) {
                    A0[i] = a0[i]; A0[i + 8] = a0[8 + i];
                    A1[i] = a1[i]; A1[i + 8] = a1[8 + i];
                }
            }
            v16bf B;
            if constexpr (SWZ) {
                B = *(const v16bf*)fp2;
                __builtin_prefetch(fp2 + 2048, 0, 1);
                fp2 += 512;
            } else {
                const float* bp = w2f + (size_t)e * SWZ_EXPERT_ELEMS
                                      + (size_t)(k0 + c * 16) * DIM + ncol;
                #pragma unroll
                for (int i = 0; i < 16; ++i) B[i] = (__bf16)bp[(size_t)i * DIM];
            }
            acc0 = __builtin_amdgcn_wmma_f32_16x16x32_bf16(false, A0, false, B,
                                                           (short)0, acc0, false, false);
            acc1 = __builtin_amdgcn_wmma_f32_16x16x32_bf16(false, A1, false, B,
                                                           (short)0, acc1, false, false);
        }
        #pragma unroll
        for (int r = 0; r < 8; ++r) {
            const int m = r + 8 * c;
            int ridx = tile * 32 + m;
            if (ridx < cnt) {
                atomicAdd(&out[(size_t)etok[base + m] * DIM + ncol],
                          ewt[base + m] * acc0[r]);
            }
            ridx += 16;
            if (ridx < cnt) {
                atomicAdd(&out[(size_t)etok[base + m + 16] * DIM + ncol],
                          ewt[base + m + 16] * acc1[r]);
            }
        }
    }
}

// ---------------------------------------------------------------------------
// 3) load-balance loss: E * sum_e (count_e/T) * (sum_scores_e/T)
// ---------------------------------------------------------------------------
__global__ void finalize_kernel(const int* __restrict__ counts,
                                const float* __restrict__ ssum,
                                float* __restrict__ out) {
    if (threadIdx.x == 0 && blockIdx.x == 0) {
        float lb = 0.0f;
        #pragma unroll
        for (int i = 0; i < NEXP; ++i)
            lb += ((float)counts[i] / (float)T_TOK) * (ssum[i] / (float)T_TOK);
        out[(size_t)T_TOK * DIM] = (float)NEXP * lb;
    }
}

// ---------------------------------------------------------------------------
extern "C" void kernel_launch(void* const* d_in, const int* in_sizes, int n_in,
                              void* d_out, int out_size, void* d_ws, size_t ws_size,
                              hipStream_t stream) {
    const float* x      = (const float*)d_in[0];
    const float* gate_w = (const float*)d_in[1];
    const float* ebias  = (const float*)d_in[2];
    const float* w1     = (const float*)d_in[3];
    const float* w3     = (const float*)d_in[4];
    const float* w2     = (const float*)d_in[5];
    float* out = (float*)d_out;

    // workspace: [stats (<=1MB)] [bf16 swizzled weights (96MB), optional]
    int*   counts = (int*)d_ws;
    float* ssum   = (float*)d_ws + NEXP;
    int*   etok   = (int*)d_ws + 2 * NEXP;
    float* ewt    = (float*)d_ws + 2 * NEXP + NEXP * T_TOK;

    __bf16* swz = (__bf16*)((char*)d_ws + SWZ_OFF_BYTES);
    __bf16* s1  = swz;
    __bf16* s3  = swz + SWZ_MAT_ELEMS;
    __bf16* s2  = swz + 2 * SWZ_MAT_ELEMS;
    const bool fast = ws_size >= SWZ_OFF_BYTES + SWZ_BYTES;

    const int n_out = T_TOK * DIM + 1;
    zero_kernel<<<(n_out + 255) / 256, 256, 0, stream>>>(out, n_out, counts, ssum);

    router_kernel<<<T_TOK, 256, 0, stream>>>(x, gate_w, ebias,
                                             counts, ssum, etok, ewt);

    const size_t lds_bytes = (size_t)2 * 32 * DIM * sizeof(__bf16); // 128 KB
    dim3 grid(NEXP, T_TOK / 32);
    if (fast) {
        swizzle_weights_kernel<<<(3u << 20) / 256, 256, 0, stream>>>(
            w1, w3, w2, s1, s3, s2);
        hipFuncSetAttribute((const void*)expert_ffn_kernel<true>,
                            hipFuncAttributeMaxDynamicSharedMemorySize,
                            (int)lds_bytes);
        expert_ffn_kernel<true><<<grid, 256, lds_bytes, stream>>>(
            x, w1, w3, w2, s1, s3, s2, counts, etok, ewt, out);
    } else {
        hipFuncSetAttribute((const void*)expert_ffn_kernel<false>,
                            hipFuncAttributeMaxDynamicSharedMemorySize,
                            (int)lds_bytes);
        expert_ffn_kernel<false><<<grid, 256, lds_bytes, stream>>>(
            x, w1, w3, w2, nullptr, nullptr, nullptr, counts, etok, ewt, out);
    }

    finalize_kernel<<<1, 32, 0, stream>>>(counts, ssum, out);
}
